// MultiHeadAttention_39195871543645
// MI455X (gfx1250) — matrix operand order repaired and checked by reading, not compiled, not run on hardware
//
#include <hip/hip_runtime.h>
#include <hip/hip_bf16.h>

#define Bb  2
#define Ss  2048
#define Dd  1024
#define Hh  16
#define DHh 64

typedef __attribute__((ext_vector_type(16))) __bf16 v16bf;
typedef __attribute__((ext_vector_type(8)))  __bf16 v8bf;
typedef __attribute__((ext_vector_type(8)))  float  v8f;
typedef __attribute__((ext_vector_type(4)))  float  v4f;

union BF16x16 { v16bf v; v8bf h[2]; unsigned short u[16]; };

__device__ __forceinline__ unsigned short f2bf(float f) {
  union { float f; unsigned u; } x; x.f = f;
  unsigned r = x.u + 0x7FFFu + ((x.u >> 16) & 1u);
  return (unsigned short)(r >> 16);
}

__device__ __forceinline__ v8f wmma_bf16(v16bf a, v16bf b, v8f c) {
  return __builtin_amdgcn_wmma_f32_16x16x32_bf16(false, a, false, b, (short)0, c, false, false);
}

// A-fragment (16x32 bf16) built from an f32 row: lane holds row M=lane&15,
// elements i -> K=kk+half*16+i
__device__ __forceinline__ v16bf load_a_f32(const float* rowptr, int kk, int half) {
  BF16x16 a;
  const float* p0 = rowptr + kk + half * 16;
  v4f x0 = *(const v4f*)(p0);
  v4f x1 = *(const v4f*)(p0 + 4);
  v4f x2 = *(const v4f*)(p0 + 8);
  v4f x3 = *(const v4f*)(p0 + 12);
#pragma unroll
  for (int j = 0; j < 4; ++j) {
    a.u[j]      = f2bf(x0[j]);
    a.u[4 + j]  = f2bf(x1[j]);
    a.u[8 + j]  = f2bf(x2[j]);
    a.u[12 + j] = f2bf(x3[j]);
  }
  return a.v;
}

// A-fragment from a bf16 (ushort) row, same element mapping, two 16B loads.
__device__ __forceinline__ v16bf load_a_bf16(const unsigned short* rowptr, int kk, int half) {
  BF16x16 a;
  a.h[0] = *(const v8bf*)(rowptr + kk + half * 16);
  a.h[1] = *(const v8bf*)(rowptr + kk + half * 16 + 8);
  return a.v;
}

// ---------------------------------------------------------------------------
// Kernel 1: convert + transpose weights to bf16.
//  WqT/WkT/WvT : [H][DH][D]  (K-contiguous for WMMA B operand)
//  WoT         : [D(out)][D(in)]
// ---------------------------------------------------------------------------
__global__ void cvt_weights_kernel(const float* __restrict__ Wq, const float* __restrict__ Wk,
                                   const float* __restrict__ Wv, const float* __restrict__ Wo,
                                   unsigned short* wqT, unsigned short* wkT,
                                   unsigned short* wvT, unsigned short* woT) {
  const int projN = Hh * Dd * DHh;         // per-projection elements
  const int total = 3 * projN + Dd * Dd;
  for (int i = blockIdx.x * blockDim.x + threadIdx.x; i < total;
       i += gridDim.x * blockDim.x) {
    if (i < 3 * projN) {
      int t = i / projN;
      int r = i - t * projN;               // r = (h*DH + e)*D + d
      int d = r % Dd;
      int he = r / Dd;
      int e = he % DHh;
      int h = he / DHh;
      const float* W = (t == 0) ? Wq : (t == 1) ? Wk : Wv;
      unsigned short* o = (t == 0) ? wqT : (t == 1) ? wkT : wvT;
      o[r] = f2bf(W[(h * Dd + d) * DHh + e]);
    } else {
      int r = i - 3 * projN;               // r = n*D + d
      int d = r % Dd;
      int n = r / Dd;
      woT[r] = f2bf(Wo[d * Dd + n]);
    }
  }
}

// ---------------------------------------------------------------------------
// Kernel 2: per-head QKV projections. One wave computes a 16x64 tile of
// q/k/v[b,h,:, :] = X[b,:,:] @ W[h] + bias[h].  32 K-steps x 4 WMMAs.
// grid = (S/16/8, H, 3*B), block = 256 (8 waves).
// ---------------------------------------------------------------------------
__global__ __launch_bounds__(256) void proj_kernel(
    const float* __restrict__ q_in, const float* __restrict__ k_in,
    const float* __restrict__ v_in,
    const unsigned short* __restrict__ wqT, const unsigned short* __restrict__ wkT,
    const unsigned short* __restrict__ wvT,
    const float* __restrict__ bq, const float* __restrict__ bk,
    const float* __restrict__ bv,
    unsigned short* __restrict__ q_bf, unsigned short* __restrict__ k_bf,
    unsigned short* __restrict__ vT_bf) {
  const int lane = threadIdx.x & 31;
  const int wave = threadIdx.x >> 5;
  const int half = lane >> 4;
  const int ln   = lane & 15;
  const int m0   = (blockIdx.x * 8 + wave) * 16;
  const int h    = blockIdx.y;
  const int mat  = blockIdx.z >> 1;        // 0=q 1=k 2=v
  const int b    = blockIdx.z & 1;

  const float* X = (mat == 0) ? q_in : (mat == 1) ? k_in : v_in;
  const unsigned short* Wt = (mat == 0) ? wqT : (mat == 1) ? wkT : wvT;
  const float* bias = (mat == 0) ? bq : (mat == 1) ? bk : bv;

  v8f acc[4];
#pragma unroll
  for (int n = 0; n < 4; ++n)
#pragma unroll
    for (int r = 0; r < 8; ++r) acc[n][r] = 0.f;

  const float* aRow = X + ((size_t)(b * Ss + (m0 + ln))) * Dd;
  const unsigned short* wRow[4];
#pragma unroll
  for (int n = 0; n < 4; ++n)
    wRow[n] = Wt + ((size_t)(h * DHh + n * 16 + ln)) * Dd;

  for (int kk = 0; kk < Dd; kk += 32) {
    v16bf a = load_a_f32(aRow, kk, half);
#pragma unroll
    for (int n = 0; n < 4; ++n) {
      v16bf bm = *(const v16bf*)(wRow[n] + kk + half * 16);
      acc[n] = wmma_bf16(a, bm, acc[n]);
    }
  }

#pragma unroll
  for (int n = 0; n < 4; ++n) {
    const int e = n * 16 + ln;
    const float bs = bias[h * DHh + e];
#pragma unroll
    for (int r = 0; r < 8; ++r) {
      const int row = m0 + r + 8 * half;
      const unsigned short val = f2bf(acc[n][r] + bs);
      if (mat == 2) {
        vT_bf[(((size_t)(b * Hh + h) * DHh + e) * Ss) + row] = val;   // [B,H,DH,S]
      } else {
        unsigned short* o = (mat == 0) ? q_bf : k_bf;
        o[(((size_t)(b * Hh + h) * Ss + row) * DHh) + e] = val;       // [B,H,S,DH]
      }
    }
  }
}

// ---------------------------------------------------------------------------
// Kernel 3: flash attention. One wave owns 16 query rows x DH=64 ctx columns.
// Streams keys 64 at a time: scores via 8 WMMAs -> LDS f32 -> online softmax
// -> bf16 probs in LDS -> 8 WMMAs into f32 ctx accumulators.
// grid = (S/16/4, H, B), block = 128 (4 waves).
// ---------------------------------------------------------------------------
#define AW 4
__global__ __launch_bounds__(128) void attn_kernel(
    const unsigned short* __restrict__ q_bf, const unsigned short* __restrict__ k_bf,
    const unsigned short* __restrict__ vT_bf, const int* __restrict__ mask,
    unsigned short* __restrict__ ctx_bf) {
  __shared__ __attribute__((aligned(32))) float          sc[AW][16 * 64];
  __shared__ __attribute__((aligned(32))) unsigned short pb[AW][16 * 64];
  __shared__ float red[AW][32];
  __shared__ float fac[AW][16];
  __shared__ float mrow[AW][16];
  __shared__ float lrow[AW][16];

  const int lane = threadIdx.x & 31;
  const int w    = threadIdx.x >> 5;
  const int half = lane >> 4;
  const int ln   = lane & 15;
  const int m0   = (blockIdx.x * AW + w) * 16;
  const int h    = blockIdx.y;
  const int b    = blockIdx.z;

  if (lane < 16) { mrow[w][lane] = -1e30f; lrow[w][lane] = 0.f; }

  v8f acc[4];
#pragma unroll
  for (int n = 0; n < 4; ++n)
#pragma unroll
    for (int r = 0; r < 8; ++r) acc[n][r] = 0.f;

  const unsigned short* qrow =
      q_bf + (((size_t)(b * Hh + h) * Ss + (m0 + ln)) * DHh);
  const v16bf qa0 = load_a_bf16(qrow, 0, half);
  const v16bf qa1 = load_a_bf16(qrow, 32, half);

  const size_t kBase = (size_t)(b * Hh + h) * Ss;   // rows of k_bf
  const size_t vBase = (size_t)(b * Hh + h) * DHh;  // rows of vT_bf

  __syncthreads();

  for (int kb = 0; kb < Ss; kb += 64) {
    // ---- scores: 4 tiles of 16 keys, K = DH = 64 (two WMMA steps) ----
#pragma unroll
    for (int t = 0; t < 4; ++t) {
      const int kcol = kb + t * 16 + ln;
      const unsigned short* krow = k_bf + (kBase + kcol) * DHh;
      v8f s;
#pragma unroll
      for (int r = 0; r < 8; ++r) s[r] = 0.f;
      s = wmma_bf16(qa0, *(const v16bf*)(krow + half * 16), s);
      s = wmma_bf16(qa1, *(const v16bf*)(krow + 32 + half * 16), s);
      const bool mz = (mask[b * Ss + kcol] == 0);
#pragma unroll
      for (int r = 0; r < 8; ++r) {
        float sv = s[r] * 0.125f;                 // 1/sqrt(64)
        if (mz) sv = -1e9f;
        sc[w][(r + 8 * half) * 64 + t * 16 + ln] = sv;
      }
    }
    __syncthreads();

    // ---- online softmax: lane handles row=ln, half of the 64 columns ----
    const int row = ln, ch = half;
    float cmax = -1e30f;
#pragma unroll
    for (int c = 0; c < 32; ++c)
      cmax = fmaxf(cmax, sc[w][row * 64 + ch * 32 + c]);
    red[w][ch * 16 + row] = cmax;
    __syncthreads();

    const float mold = mrow[w][row];
    const float mnew = fmaxf(mold, fmaxf(red[w][row], red[w][16 + row]));
    const float f    = __expf(mold - mnew);
    float csum = 0.f;
#pragma unroll
    for (int c = 0; c < 32; ++c) {
      const float p = __expf(sc[w][row * 64 + ch * 32 + c] - mnew);
      pb[w][row * 64 + ch * 32 + c] = f2bf(p);
      csum += p;
    }
    red[w][ch * 16 + row] = csum;
    if (ch == 0) fac[w][row] = f;
    __syncthreads();

    if (ch == 0) {
      lrow[w][row] = lrow[w][row] * f + red[w][row] + red[w][16 + row];
      mrow[w][row] = mnew;
    }

    // ---- rescale accumulators by exp(mold - mnew) per row ----
#pragma unroll
    for (int r = 0; r < 8; ++r) {
      const float fr = fac[w][r + 8 * half];
      acc[0][r] *= fr; acc[1][r] *= fr; acc[2][r] *= fr; acc[3][r] *= fr;
    }

    // ---- ctx += P(16x64) @ V(64x64) : 2 K-steps x 4 N-tiles ----
#pragma unroll
    for (int kkh = 0; kkh < 2; ++kkh) {
      const int kk = kkh * 32;
      const v16bf pa = load_a_bf16(&pb[w][ln * 64], kk, half);
#pragma unroll
      for (int n = 0; n < 4; ++n) {
        const unsigned short* vrow =
            vT_bf + (vBase + n * 16 + ln) * Ss + kb + kk + half * 16;
        acc[n] = wmma_bf16(pa, *(const v16bf*)vrow, acc[n]);
      }
    }
  }
  __syncthreads();

  // ---- normalize and store ctx as bf16 in [B,S,D] with col = h*DH + e ----
#pragma unroll
  for (int r = 0; r < 8; ++r) {
    const float inv = 1.0f / lrow[w][r + 8 * half];
    const int rowi = m0 + r + 8 * half;
#pragma unroll
    for (int n = 0; n < 4; ++n) {
      const int col = h * DHh + n * 16 + ln;
      ctx_bf[((size_t)(b * Ss + rowi)) * Dd + col] = f2bf(acc[n][r] * inv);
    }
  }
}

// ---------------------------------------------------------------------------
// Kernel 4: output projection  out = ctx @ Wo + bo  (4096 x 1024 x 1024).
// grid = (B*S/16/8, D/64), block = 256 (8 waves), wave tile 16x64.
// ---------------------------------------------------------------------------
__global__ __launch_bounds__(256) void outproj_kernel(
    const unsigned short* __restrict__ ctx_bf, const unsigned short* __restrict__ woT,
    const float* __restrict__ bo, float* __restrict__ out) {
  const int lane = threadIdx.x & 31;
  const int wave = threadIdx.x >> 5;
  const int half = lane >> 4;
  const int ln   = lane & 15;
  const int m0   = (blockIdx.x * 8 + wave) * 16;   // rows over B*S
  const int n0   = blockIdx.y * 64;

  v8f acc[4];
#pragma unroll
  for (int n = 0; n < 4; ++n)
#pragma unroll
    for (int r = 0; r < 8; ++r) acc[n][r] = 0.f;

  const unsigned short* arow = ctx_bf + ((size_t)(m0 + ln)) * Dd;
  const unsigned short* brow[4];
#pragma unroll
  for (int n = 0; n < 4; ++n)
    brow[n] = woT + ((size_t)(n0 + n * 16 + ln)) * Dd;

  for (int kk = 0; kk < Dd; kk += 32) {
    const v16bf a = load_a_bf16(arow, kk, half);
#pragma unroll
    for (int n = 0; n < 4; ++n) {
      const v16bf bm = *(const v16bf*)(brow[n] + kk + half * 16);
      acc[n] = wmma_bf16(a, bm, acc[n]);
    }
  }

#pragma unroll
  for (int n = 0; n < 4; ++n) {
    const int col = n0 + n * 16 + ln;
    const float bs = bo[col];
#pragma unroll
    for (int r = 0; r < 8; ++r) {
      const int row = m0 + r + 8 * half;
      out[(size_t)row * Dd + col] = acc[n][r] + bs;
    }
  }
}

// ---------------------------------------------------------------------------
extern "C" void kernel_launch(void* const* d_in, const int* in_sizes, int n_in,
                              void* d_out, int out_size, void* d_ws, size_t ws_size,
                              hipStream_t stream) {
  const float* query = (const float*)d_in[0];
  const float* key_  = (const float*)d_in[1];
  const float* value = (const float*)d_in[2];
  const int*   mask  = (const int*)d_in[3];
  const float* Wq = (const float*)d_in[4];
  const float* bq = (const float*)d_in[5];
  const float* Wk = (const float*)d_in[6];
  const float* bk = (const float*)d_in[7];
  const float* Wv = (const float*)d_in[8];
  const float* bv = (const float*)d_in[9];
  const float* Wo = (const float*)d_in[10];
  const float* bo = (const float*)d_in[11];
  float* out = (float*)d_out;

  unsigned short* ws = (unsigned short*)d_ws;
  const size_t WSZ = (size_t)Hh * DHh * Dd;      // 1,048,576 elems per W
  const size_t OSZ = (size_t)Dd * Dd;            // Wo
  const size_t QSZ = (size_t)Bb * Hh * Ss * DHh; // 4,194,304 elems

  unsigned short* wqT   = ws;
  unsigned short* wkT   = wqT + WSZ;
  unsigned short* wvT   = wkT + WSZ;
  unsigned short* woT   = wvT + WSZ;
  unsigned short* qbf   = woT + OSZ;
  unsigned short* kbf   = qbf + QSZ;
  unsigned short* vTbf  = kbf + QSZ;
  unsigned short* ctxbf = vTbf + QSZ;            // total ~40 MB of d_ws

  cvt_weights_kernel<<<1024, 256, 0, stream>>>(Wq, Wk, Wv, Wo, wqT, wkT, wvT, woT);

  proj_kernel<<<dim3(Ss / 16 / 8, Hh, 3 * Bb), 256, 0, stream>>>(
      query, key_, value, wqT, wkT, wvT, bq, bk, bv, qbf, kbf, vTbf);

  attn_kernel<<<dim3(Ss / 16 / AW, Hh, Bb), 32 * AW, 0, stream>>>(
      qbf, kbf, vTbf, mask, ctxbf);

  outproj_kernel<<<dim3((Bb * Ss) / 16 / 8, Dd / 64, 1), 256, 0, stream>>>(
      ctxbf, woT, bo, out);
}
